// GraphSummary_7215545057986
// MI455X (gfx1250) — compile-verified, matrix operand order checked by reading, not run on hardware
//
#include <hip/hip_runtime.h>
#include <hip/hip_bf16.h>

typedef __attribute__((ext_vector_type(16))) __bf16 v16bf;
typedef __attribute__((ext_vector_type(8)))  float  v8f;

#define D_IN 6
#define H    128
#define B_SZ 128
#define N_SZ 64
#define T_SZ 64
#define BN   (B_SZ*N_SZ)        // 8192
#define ROWS (BN*T_SZ)          // 524288
#define DM   134
#define KPAD 160                // 5 k-chunks of 32
#define H3   384

// ---- workspace layout (bytes) ----
#define WS_WFUSED 0                              // 6*384 f32
#define WS_BFUSED (WS_WFUSED + D_IN*H3*4)        // 384 f32
#define WS_WHH    (WS_BFUSED + H3*4)             // 384*128 bf16
#define WS_WG1    (WS_WHH + H3*H*2)              // 160*128 bf16 (zero-padded)
#define WS_WN1    (WS_WG1 + KPAD*H*2)            // 160*128 bf16
#define WS_WN2    (WS_WN1 + KPAD*H*2)            // 128*128 bf16
#define WS_HLAST  (WS_WN2 + H*H*2)               // 8192*128 f32
#define WS_PART   (WS_HLAST + BN*H*4)            // 128*32*130 f32

// ---------- WMMA fragment helpers (layouts per ISA 7.12.2, wave32) ----------
__device__ __forceinline__ int akof(int v, int hi) {
  // VGPR v holds K pair base: v0..3 -> K 0..7 (+hi*8), v4..7 -> K 16..23 (+hi*8)
  return ((v & 4) << 2) | ((v & 3) << 1) | (hi << 3);
}
// fragment whose K values are contiguous in memory at p[k]
__device__ __forceinline__ v16bf frag_k_contig(const __bf16* p, int hi) {
  v16bf f;
#pragma unroll
  for (int v = 0; v < 8; ++v) {
    int k = akof(v, hi);
    f[2*v]   = p[k];
    f[2*v+1] = p[k + 1];
  }
  return f;
}
// fragment whose K values stride by ldn in memory (B from row-major [k][n])
__device__ __forceinline__ v16bf frag_k_strided(const __bf16* p, int ldn, int hi) {
  v16bf f;
#pragma unroll
  for (int v = 0; v < 8; ++v) {
    int k = akof(v, hi);
    f[2*v]   = p[(size_t)k * ldn];
    f[2*v+1] = p[(size_t)(k + 1) * ldn];
  }
  return f;
}
__device__ __forceinline__ v8f wmma_bf16(v16bf a, v16bf b, v8f c) {
  return __builtin_amdgcn_wmma_f32_16x16x32_bf16(false, a, false, b, (short)0, c,
                                                 false, false);
}
__device__ __forceinline__ float sigmoidf_(float x) { return 1.f / (1.f + __expf(-x)); }
__device__ __forceinline__ float tanhf_(float x)    { float e = __expf(2.f*x); return (e-1.f)/(e+1.f); }
__device__ __forceinline__ float siluf_(float x)    { return x * sigmoidf_(x); }

// ---------- kernel A1: fold embedding into the GRU input projection ----------
// W_fused[d][j] = sum_k W_emb[d][k] * W_ih[j][k];  b_fused[j] = b_emb@W_ih.T + b_ih
__global__ void prep_fused_kernel(const float* __restrict__ W_emb,
                                  const float* __restrict__ b_emb,
                                  const float* __restrict__ W_ih,
                                  const float* __restrict__ b_ih,
                                  float* __restrict__ Wf, float* __restrict__ bf) {
  int j = threadIdx.x;
  if (j >= H3) return;
  const float* wrow = W_ih + (size_t)j * H;
  float accb = b_ih[j];
  for (int k = 0; k < H; ++k) accb += b_emb[k] * wrow[k];
  bf[j] = accb;
  for (int d = 0; d < D_IN; ++d) {
    const float* we = W_emb + (size_t)d * H;
    float acc = 0.f;
    for (int k = 0; k < H; ++k) acc += we[k] * wrow[k];
    Wf[(size_t)d * H3 + j] = acc;
  }
}

// ---------- kernel A2: f32 -> bf16 with zero tail-padding ----------
__global__ void convert_pad_kernel(const float* __restrict__ src, __bf16* __restrict__ dst,
                                   int srcN, int dstN) {
  int i = blockIdx.x * blockDim.x + threadIdx.x;
  if (i < dstN) dst[i] = (i < srcN) ? (__bf16)src[i] : (__bf16)0.f;
}

// ---------- kernel C: GRU recurrence, h in LDS, W_hh resident in VGPRs ----------
__global__ void __launch_bounds__(256) gru_kernel(
    const float* __restrict__ x, const float* __restrict__ Wf,
    const float* __restrict__ bf, const __bf16* __restrict__ whh,
    const float* __restrict__ b_hh, float* __restrict__ hlast) {
  __shared__ __bf16 hbuf[H * H];                 // 128 seqs x 128 hidden, bf16 (32 KB)
  const int tid = threadIdx.x;
  const int w = tid >> 5, lane = tid & 31;
  const int l = lane & 15, hi = lane >> 4;
  const int seq0 = blockIdx.x * 128;
  const int cl = 16 * w + l;                     // this lane's hidden column

  for (int i = tid; i < H * H; i += 256) hbuf[i] = (__bf16)0.f;

  // B fragments of W_hh (row-major [j][k]; B(k,n)=whh[j0+n][k] -> contiguous k)
  v16bf bR[4], bZ[4], bN[4];
#pragma unroll
  for (int kc = 0; kc < 4; ++kc) {
    bR[kc] = frag_k_contig(whh + (size_t)(cl      ) * H + kc * 32, hi);
    bZ[kc] = frag_k_contig(whh + (size_t)(cl + 128) * H + kc * 32, hi);
    bN[kc] = frag_k_contig(whh + (size_t)(cl + 256) * H + kc * 32, hi);
  }
  float wfR[6], wfZ[6], wfN[6];
#pragma unroll
  for (int d = 0; d < D_IN; ++d) {
    wfR[d] = Wf[d * H3 + cl];
    wfZ[d] = Wf[d * H3 + cl + 128];
    wfN[d] = Wf[d * H3 + cl + 256];
  }
  const float bfR = bf[cl], bfZ = bf[cl + 128], bfN = bf[cl + 256];
  const float bhR = b_hh[cl], bhZ = b_hh[cl + 128], bhN = b_hh[cl + 256];
  __syncthreads();

  float hnew[64];
  for (int t = 0; t < T_SZ; ++t) {
#pragma unroll
    for (int mt = 0; mt < 8; ++mt) {
      v8f aR = {}, aZ = {}, aN = {};
#pragma unroll
      for (int kc = 0; kc < 4; ++kc) {
        v16bf a = frag_k_contig(&hbuf[(mt * 16 + l) * H + kc * 32], hi);
        aR = wmma_bf16(a, bR[kc], aR);
        aZ = wmma_bf16(a, bZ[kc], aZ);
        aN = wmma_bf16(a, bN[kc], aN);
      }
#pragma unroll
      for (int r = 0; r < 8; ++r) {
        int m = mt * 16 + r + 8 * hi;
        const float* xr = x + ((size_t)(seq0 + m) * T_SZ + t) * D_IN;
        float xpR = bfR, xpZ = bfZ, xpN = bfN;
#pragma unroll
        for (int d = 0; d < D_IN; ++d) {
          float xv = xr[d];
          xpR += xv * wfR[d]; xpZ += xv * wfZ[d]; xpN += xv * wfN[d];
        }
        float ho = (float)hbuf[m * H + cl];
        float rg = sigmoidf_(xpR + aR[r] + bhR);
        float zg = sigmoidf_(xpZ + aZ[r] + bhZ);
        float ng = tanhf_(xpN + rg * (aN[r] + bhN));
        hnew[mt * 8 + r] = (1.f - zg) * ng + zg * ho;
      }
    }
    __syncthreads();                              // all reads of h done
#pragma unroll
    for (int mt = 0; mt < 8; ++mt)
#pragma unroll
      for (int r = 0; r < 8; ++r)
        hbuf[(mt * 16 + r + 8 * hi) * H + cl] = (__bf16)hnew[mt * 8 + r];
    __syncthreads();                              // all writes done
  }
  for (int i = tid; i < H * H; i += 256) {
    int m = i >> 7, k = i & 127;
    hlast[(size_t)(seq0 + m) * H + k] = (float)hbuf[i];
  }
}

// ---------- kernel D: fused head (2 GEMMs + silu + gate + online softmax partial) ----------
__global__ void __launch_bounds__(256) head_kernel(
    const float* __restrict__ x, const float* __restrict__ hlast,
    const __bf16* __restrict__ wg1p, const float* __restrict__ bg1,
    const float* __restrict__ Wg2, const float* __restrict__ bg2,
    const __bf16* __restrict__ wn1p, const float* __restrict__ bn1,
    const __bf16* __restrict__ wn2p, const float* __restrict__ bn2,
    float* __restrict__ part) {
  __shared__ __bf16 feats[128 * KPAD];           // 40 KB; later reused for silu(v1)
  __shared__ float gates[128];
  __shared__ float wsum[128];
  const int tid = threadIdx.x;
  const int w = tid >> 5, lane = tid & 31;
  const int l = lane & 15, hi = lane >> 4;
  const int row0 = blockIdx.x * 128;
  const int bseg = row0 >> 12;                   // segment = row / 4096
  const int wgInSeg = (row0 >> 7) & 31;
  const int m0 = 16 * w;                         // wave owns rows m0..m0+15

  if (tid < 128) wsum[tid] = 0.f;
  // stage feats = [x(6) | h(128)] + posenc, bf16, k padded to 160 with zeros
  for (int i = tid; i < 128 * KPAD; i += 256) {
    int m = i / KPAD, k = i - m * KPAD;
    int row = row0 + m;
    int t = row & (T_SZ - 1);
    int seq = row >> 6;
    float v = 0.f;
    if (k < DM) {
      float arg = (float)t * __expf(-9.2103404f * (float)(k & ~1) / 134.f);
      float pe = (k & 1) ? __cosf(arg) : __sinf(arg);
      float base = (k < D_IN) ? x[(size_t)row * D_IN + k]
                              : hlast[(size_t)seq * H + (k - D_IN)];
      v = base + pe;
    }
    feats[i] = (__bf16)v;
  }
  __syncthreads();

  // A fragments of feats for this wave's rows (invariant over output tiles)
  v16bf af[5];
#pragma unroll
  for (int kc = 0; kc < 5; ++kc)
    af[kc] = frag_k_contig(&feats[(m0 + l) * KPAD + kc * 32], hi);

  float pg[8];
#pragma unroll
  for (int r = 0; r < 8; ++r) pg[r] = 0.f;
  float vreg[64];
#pragma unroll
  for (int nt = 0; nt < 8; ++nt) {
    int c = nt * 16 + l;
    v8f aG = {}, aV = {};
#pragma unroll
    for (int kc = 0; kc < 5; ++kc) {
      v16bf bg = frag_k_strided(wg1p + (size_t)(kc * 32) * H + c, H, hi);
      v16bf bv = frag_k_strided(wn1p + (size_t)(kc * 32) * H + c, H, hi);
      aG = wmma_bf16(af[kc], bg, aG);
      aV = wmma_bf16(af[kc], bv, aV);
    }
    float bg1c = bg1[c], bn1c = bn1[c], wg2c = Wg2[c];
#pragma unroll
    for (int r = 0; r < 8; ++r) {
      pg[r] += siluf_(aG[r] + bg1c) * wg2c;
      vreg[nt * 8 + r] = siluf_(aV[r] + bn1c);
    }
  }
  // gate[m] = reduce over hidden (16 lanes per half) + bg2
  float bg2v = bg2[0];
#pragma unroll
  for (int r = 0; r < 8; ++r) {
    float p = pg[r];
    p += __shfl_xor(p, 1);
    p += __shfl_xor(p, 2);
    p += __shfl_xor(p, 4);
    p += __shfl_xor(p, 8);
    if (l == 0) gates[m0 + r + 8 * hi] = p + bg2v;
  }
  // stash silu(v1) over this wave's own feats rows (wave-private, no barrier)
#pragma unroll
  for (int nt = 0; nt < 8; ++nt)
#pragma unroll
    for (int r = 0; r < 8; ++r)
      feats[(m0 + r + 8 * hi) * KPAD + nt * 16 + l] = (__bf16)vreg[nt * 8 + r];

  // GEMM2: val = silu(v1) @ Wn2 + bn2
  v16bf av2[4];
#pragma unroll
  for (int kc = 0; kc < 4; ++kc)
    av2[kc] = frag_k_contig(&feats[(m0 + l) * KPAD + kc * 32], hi);
  float oreg[64];
#pragma unroll
  for (int nt = 0; nt < 8; ++nt) {
    int c = nt * 16 + l;
    v8f aO = {};
#pragma unroll
    for (int kc = 0; kc < 4; ++kc) {
      v16bf b = frag_k_strided(wn2p + (size_t)(kc * 32) * H + c, H, hi);
      aO = wmma_bf16(av2[kc], b, aO);
    }
    float bn2c = bn2[c];
#pragma unroll
    for (int r = 0; r < 8; ++r) oreg[nt * 8 + r] = aO[r] + bn2c;
  }
  __syncthreads();                               // gates complete

  float mloc = -3.4e38f;
  for (int i = 0; i < 128; ++i) mloc = fmaxf(mloc, gates[i]);
  float s = 0.f;
  for (int i = 0; i < 128; ++i) s += __expf(gates[i] - mloc);
  float em[8];
#pragma unroll
  for (int r = 0; r < 8; ++r) em[r] = __expf(gates[m0 + r + 8 * hi] - mloc);
#pragma unroll
  for (int nt = 0; nt < 8; ++nt) {
    float pv = 0.f;
#pragma unroll
    for (int r = 0; r < 8; ++r) pv += em[r] * oreg[nt * 8 + r];
    pv += __shfl_xor(pv, 16);                    // merge the two row-halves
    if (hi == 0) atomicAdd(&wsum[nt * 16 + l], pv);
  }
  __syncthreads();
  float* prec = part + (size_t)(bseg * 32 + wgInSeg) * 130;
  if (tid == 0) { prec[0] = mloc; prec[1] = s; }
  if (tid < 128) prec[2 + tid] = wsum[tid];
}

// ---------- kernel E: merge 32 softmax partials per segment ----------
__global__ void combine_kernel(const float* __restrict__ part, float* __restrict__ out) {
  int b = blockIdx.x, c = threadIdx.x;
  const float* pb = part + (size_t)b * 32 * 130;
  float M = -3.4e38f;
  for (int p = 0; p < 32; ++p) M = fmaxf(M, pb[p * 130]);
  float denom = 0.f, acc = 0.f;
  for (int p = 0; p < 32; ++p) {
    float sc = __expf(pb[p * 130] - M);
    denom += sc * pb[p * 130 + 1];
    acc   += sc * pb[p * 130 + 2 + c];
  }
  out[(size_t)b * H + c] = acc / denom;
}

extern "C" void kernel_launch(void* const* d_in, const int* in_sizes, int n_in,
                              void* d_out, int out_size, void* d_ws, size_t ws_size,
                              hipStream_t stream) {
  (void)in_sizes; (void)n_in; (void)out_size; (void)ws_size;
  const float* x     = (const float*)d_in[0];
  const float* W_emb = (const float*)d_in[1];
  const float* b_emb = (const float*)d_in[2];
  const float* W_ih  = (const float*)d_in[3];
  const float* W_hh  = (const float*)d_in[4];
  const float* b_ih  = (const float*)d_in[5];
  const float* b_hh  = (const float*)d_in[6];
  const float* Wg1   = (const float*)d_in[7];
  const float* bg1   = (const float*)d_in[8];
  const float* Wg2   = (const float*)d_in[9];
  const float* bg2   = (const float*)d_in[10];
  const float* Wn1   = (const float*)d_in[11];
  const float* bn1   = (const float*)d_in[12];
  const float* Wn2   = (const float*)d_in[13];
  const float* bn2   = (const float*)d_in[14];

  char* ws = (char*)d_ws;
  float*  Wf    = (float*)(ws + WS_WFUSED);
  float*  bfv   = (float*)(ws + WS_BFUSED);
  __bf16* whh   = (__bf16*)(ws + WS_WHH);
  __bf16* wg1p  = (__bf16*)(ws + WS_WG1);
  __bf16* wn1p  = (__bf16*)(ws + WS_WN1);
  __bf16* wn2p  = (__bf16*)(ws + WS_WN2);
  float*  hlast = (float*)(ws + WS_HLAST);
  float*  part  = (float*)(ws + WS_PART);

  prep_fused_kernel<<<1, H3, 0, stream>>>(W_emb, b_emb, W_ih, b_ih, Wf, bfv);
  convert_pad_kernel<<<(H3 * H + 255) / 256, 256, 0, stream>>>(W_hh, whh, H3 * H, H3 * H);
  convert_pad_kernel<<<(KPAD * H + 255) / 256, 256, 0, stream>>>(Wg1, wg1p, DM * H, KPAD * H);
  convert_pad_kernel<<<(KPAD * H + 255) / 256, 256, 0, stream>>>(Wn1, wn1p, DM * H, KPAD * H);
  convert_pad_kernel<<<(H * H + 255) / 256, 256, 0, stream>>>(Wn2, wn2p, H * H, H * H);

  gru_kernel<<<BN / 128, 256, 0, stream>>>(x, Wf, bfv, whh, b_hh, hlast);
  head_kernel<<<ROWS / 128, 256, 0, stream>>>(x, hlast, wg1p, bg1, Wg2, bg2,
                                              wn1p, bn1, wn2p, bn2, part);
  combine_kernel<<<B_SZ, H, 0, stream>>>(part, (float*)d_out);
}
